// YuanAttention_32684701122989
// MI455X (gfx1250) — matrix operand order repaired and checked by reading, not compiled, not run on hardware
//
#include <hip/hip_runtime.h>
#include <math.h>

// ---------------- types / constants ----------------
typedef __attribute__((ext_vector_type(16))) __bf16 v16bf;
typedef __attribute__((ext_vector_type(8)))  float  v8f;
typedef __attribute__((ext_vector_type(4)))  unsigned int tdm_v4u;
typedef __attribute__((ext_vector_type(8)))  int          tdm_v8i;
typedef __attribute__((ext_vector_type(4)))  int          tdm_v4i;

#define WMMA_BF16(A_, B_, C_) \
  __builtin_amdgcn_wmma_f32_16x16x32_bf16(false, (A_), false, (B_), (short)0, (C_), false, false)

#define B_SZ   2
#define S_SZ   2048
#define H_SZ   2048
#define NH_SZ  32
#define HD_SZ  64
#define T_SZ   (B_SZ * S_SZ)      // 4096 tokens

#define GKC    4                  // K-steps (of 32) staged per TDM chunk

#define MSCALE 1.0693147180559945f  // 0.1*ln(2)+1 (SCALE=2)

#if defined(__AMDGCN__) && __has_builtin(__builtin_amdgcn_tensor_load_to_lds)
#define USE_TDM 1
#else
#define USE_TDM 0
#endif

// YaRN-corrected inverse frequency for freq index fi in [0,32)
__device__ __forceinline__ float yarn_invfreq(int fi) {
  const float lgBASE = 9.210340371976184f;   // ln(10000)
  const float TWO_PI = 6.283185307179586f;
  float inv = __expf(-(2.0f * (float)fi / 64.0f) * lgBASE);
  float low  = floorf(64.0f * logf(8192.0f / (32.0f * TWO_PI)) / (2.0f * lgBASE));
  float high = ceilf (64.0f * logf(8192.0f / ( 1.0f * TWO_PI)) / (2.0f * lgBASE));
  low = fmaxf(low, 0.0f); high = fminf(high, 63.0f);
  float hi = (low == high) ? high + 0.001f : high;
  float ramp = ((float)fi - low) / (hi - low);
  ramp = fminf(fmaxf(ramp, 0.0f), 1.0f);
  return inv / (1.0f + ramp);   // (1-mask)*SCALE + mask == 1+ramp for SCALE=2
}

#if USE_TDM
// 2-D TDM load: tile_x elements (data_size each) x tile_y rows, row stride
// stride_elems, from gaddr into LDS byte offset lds_addr.  D# per ISA ch.8.
// data_size_code: 0=1B,1=2B,2=4B,3=8B
// 6-arg builtin form: (v4u g0, v8i g1, v4i g2, v4i g3, v8i aux, i32 cpol)
__device__ __forceinline__ void tdm_load_2d(unsigned lds_addr, const void* gaddr,
                                            unsigned tile_x, unsigned tile_y,
                                            unsigned long long stride_elems,
                                            unsigned data_size_code) {
  unsigned long long ga = (unsigned long long)gaddr;
  tdm_v4u g0;
  g0[0] = 1u;                                                  // count=1 (valid D#)
  g0[1] = lds_addr;                                            // lds_addr[31:0]
  g0[2] = (unsigned)ga;                                        // global_addr[31:0]
  g0[3] = (unsigned)((ga >> 32) & 0x01FFFFFFu) | 0x80000000u;  // addr[56:32] | type=2
  tdm_v8i g1;
  g1[0] = (int)(data_size_code << 16);                         // wg_mask=0, data_size
  g1[1] = (int)((tile_x & 0xFFFFu) << 16);                     // tensor_dim0[15:0]
  g1[2] = (int)((tile_x >> 16) | ((tile_y & 0xFFFFu) << 16));  // td0 hi | td1 lo
  g1[3] = (int)((tile_y >> 16) | (tile_x << 16));              // td1 hi | tile_dim0
  g1[4] = (int)(tile_y & 0xFFFFu);                             // tile_dim1 | tile_dim2=0
  g1[5] = (int)(stride_elems & 0xFFFFFFFFull);                 // dim0_stride[31:0]
  g1[6] = (int)((stride_elems >> 32) & 0xFFFFull);             // dim0_stride[47:32]
  g1[7] = 0;
  tdm_v4i z4;
  z4[0] = 0; z4[1] = 0; z4[2] = 0; z4[3] = 0;
  tdm_v8i z8;
  z8[0] = 0; z8[1] = 0; z8[2] = 0; z8[3] = 0;
  z8[4] = 0; z8[5] = 0; z8[6] = 0; z8[7] = 0;
  __builtin_amdgcn_tensor_load_to_lds(g0, g1, z4, z4, z8, 0);
}
#endif

// ---------------- weight pre-swizzle into WMMA B-fragment layout ----------------
// B-frag (32xK x 16xN bf16 tile): lane L -> column N=L%16, K rows (L<16?0:16)+e.
// Storage: idx = ((tn*Ktiles + tk)*32 + lane)*16 + e   (32B contiguous per lane)
// mode 0: plain row-major W[N][K]
// mode 1: conv "cat": col c -> tap=c/Ic, i=c%Ic, src[(n*Ic+i)*2 + tap]
__global__ void pack_b(const float* __restrict__ W, int Ntiles, int Ktiles,
                       int K, int mode, int Ic,
                       __bf16* __restrict__ Bhi, __bf16* __restrict__ Blo) {
  int gid = blockIdx.x * blockDim.x + threadIdx.x;
  int total = Ntiles * Ktiles * 32;
  if (gid >= total) return;
  int lane = gid & 31;
  int tile = gid >> 5;
  int tk = tile % Ktiles;
  int tn = tile / Ktiles;
  int n  = tn * 16 + (lane & 15);
  int kb = tk * 32 + ((lane & 16) ? 16 : 0);
  size_t out = (size_t)gid * 16;
  #pragma unroll 4
  for (int e = 0; e < 16; ++e) {
    int c = kb + e;
    float v;
    if (mode == 0) v = W[(size_t)n * K + c];
    else { int tap = c / Ic; int i = c - tap * Ic; v = W[((size_t)n * Ic + i) * 2 + tap]; }
    __bf16 h = (__bf16)v;
    Bhi[out + e] = h;
    Blo[out + e] = (__bf16)(v - (float)h);
  }
}

// ---------------- build [x[s-1] | x[s]] concatenated rows ----------------
__global__ void build_cat(const float* __restrict__ src, const float* __restrict__ cache,
                          int K0, float* __restrict__ dst) {
  long idx = (long)blockIdx.x * blockDim.x + threadIdx.x;
  long total = (long)T_SZ * 2 * K0;
  if (idx >= total) return;
  int c = (int)(idx % (2 * K0));
  int t = (int)(idx / (2 * K0));
  int s = t & (S_SZ - 1);
  int b = t >> 11;
  float v;
  if (c < K0) v = (s == 0) ? cache[(size_t)b * K0 + c] : src[(size_t)(t - 1) * K0 + c];
  else        v = src[(size_t)t * K0 + (c - K0)];
  dst[idx] = v;
}

// ---------------- GEMM: C(MxN) = A(MxK,f32) * W(NxK)^T via bf16x3 WMMA ----------------
// 4-wave workgroup -> 16(M) x 256(N) tile; wave w covers N columns [w*64, w*64+64).
// A tile (16 x GKC*32 fp32) staged into LDS by the Tensor Data Mover, double
// buffered on TENSORcnt; the 4 waves each convert 1/GKC of the chunk into a
// shared LDS hi/lo A-fragment buffer (bf16x3 split), then all waves consume it.
// epilogue modes: 0 = row-major store (+bias)
//                 1 = store V transposed per head: vT[((b*NH+nh)*HD+d)*S + s]
//                 2 = fused YaRN RoPE; q (c<64, scaled 0.125) -> out0, k -> out1
__global__ void __launch_bounds__(128)
gemm_bf16x3(const float* __restrict__ A, int lda, int K,
            const __bf16* __restrict__ Bhi, const __bf16* __restrict__ Blo,
            const float* __restrict__ bias, int mode,
            float* __restrict__ out0, int ldc,
            float* __restrict__ out1, const int* __restrict__ pos) {
  const int tid  = threadIdx.x;
  const int lane = tid & 31;
  const int wid  = tid >> 5;
  const int m0 = blockIdx.y * 16;
  const int n0 = blockIdx.x * 256 + wid * 64;
  const int ktiles = K >> 5;
  const int nchunk = ktiles / GKC;
  const int l15 = lane & 15;
  const int base8 = (lane & 16) ? 8 : 0;

  __shared__ __attribute__((aligned(16))) float  Araw[2][16 * GKC * 32];
  __shared__ __attribute__((aligned(32))) __bf16 PackAhi[GKC][32 * 16];
  __shared__ __attribute__((aligned(32))) __bf16 PackAlo[GKC][32 * 16];

  v8f acc[4];
  #pragma unroll
  for (int j = 0; j < 4; ++j) acc[j] = (v8f)(0.0f);

#if USE_TDM
  if (wid == 0) {
    tdm_load_2d((unsigned)(unsigned long long)(void*)&Araw[0][0],
                A + (size_t)m0 * lda, GKC * 32, 16, (unsigned long long)lda, 2u);
  }
#endif

  for (int c = 0; c < nchunk; ++c) {
    const int dbuf = c & 1;
#if USE_TDM
    if (wid == 0) {
      if (c + 1 < nchunk) {
        tdm_load_2d((unsigned)(unsigned long long)(void*)&Araw[(c + 1) & 1][0],
                    A + (size_t)m0 * lda + (size_t)(c + 1) * GKC * 32,
                    GKC * 32, 16, (unsigned long long)lda, 2u);
        __builtin_amdgcn_s_wait_tensorcnt(1);
      } else {
        __builtin_amdgcn_s_wait_tensorcnt(0);
      }
    }
#else
    for (int i = tid; i < 16 * GKC * 32; i += 128) {
      int y = i / (GKC * 32), xx = i - y * (GKC * 32);
      Araw[dbuf][i] = A[(size_t)(m0 + y) * lda + (size_t)c * GKC * 32 + xx];
    }
#endif
    __syncthreads();

    // distributed bf16x3 conversion: wave w converts local k-step w
    {
      const float* src = &Araw[dbuf][l15 * (GKC * 32) + wid * 32 + (lane & 16)];
      v16bf ahi, alo;
      #pragma unroll
      for (int i = 0; i < 8; ++i) {
        float f0 = src[i];
        float f1 = src[8 + i];
        __bf16 h0 = (__bf16)f0, h1 = (__bf16)f1;
        ahi[i] = h0;        ahi[8 + i] = h1;
        alo[i] = (__bf16)(f0 - (float)h0);
        alo[8 + i] = (__bf16)(f1 - (float)h1);
      }
      *(v16bf*)&PackAhi[wid][lane * 16] = ahi;
      *(v16bf*)&PackAlo[wid][lane * 16] = alo;
    }
    __syncthreads();

    #pragma unroll
    for (int ktl = 0; ktl < GKC; ++ktl) {
      const int kt = c * GKC + ktl;
      v16bf ahi = *(const v16bf*)&PackAhi[ktl][lane * 16];
      v16bf alo = *(const v16bf*)&PackAlo[ktl][lane * 16];
      size_t bbase = (((size_t)(n0 >> 4) * ktiles + kt) * 32 + lane) * 16;
      #pragma unroll
      for (int j = 0; j < 4; ++j) {
        size_t off = bbase + (size_t)j * ktiles * 512;
        v16bf bhi = *(const v16bf*)(Bhi + off);
        v16bf blo = *(const v16bf*)(Blo + off);
        acc[j] = WMMA_BF16(ahi, bhi, acc[j]);
        acc[j] = WMMA_BF16(ahi, blo, acc[j]);
        acc[j] = WMMA_BF16(alo, bhi, acc[j]);
      }
    }
    __syncthreads();
  }

  const int rb = m0 + base8;   // C frag: lane -> N=l15, rows rb..rb+7
  if (mode == 0) {
    #pragma unroll
    for (int j = 0; j < 4; ++j) {
      int n = n0 + j * 16 + l15;
      float bv = bias ? bias[n] : 0.0f;
      #pragma unroll
      for (int r = 0; r < 8; ++r)
        out0[(size_t)(rb + r) * ldc + n] = acc[j][r] + bv;
    }
  } else if (mode == 1) {
    #pragma unroll
    for (int j = 0; j < 4; ++j) {
      int n  = n0 + j * 16 + l15;
      int nh = n >> 6, d = n & 63;
      #pragma unroll
      for (int r = 0; r < 8; ++r) {
        int t = rb + r;
        int b = t >> 11, s = t & (S_SZ - 1);
        out0[(((size_t)b * NH_SZ + nh) * HD_SZ + d) * S_SZ + s] = acc[j][r];
      }
    }
  } else {  // mode 2: fused YaRN RoPE; rotate-half partner is frag j^2, same lane/row
    #pragma unroll
    for (int r = 0; r < 8; ++r) {
      int t = rb + r;
      float pv = (float)pos[t];
      #pragma unroll
      for (int j = 0; j < 4; ++j) {
        int n  = n0 + j * 16 + l15;
        int nh = n >> 7;
        int c  = n & 127;
        int d  = c & 63;
        int fi = d & 31;
        float th = pv * yarn_invfreq(fi);
        float cs = cosf(th) * MSCALE;
        float sn = sinf(th) * MSCALE;
        float x  = acc[j][r];
        float pr = acc[j ^ 2][r];
        float o  = x * cs + ((d < 32) ? -pr : pr) * sn;
        if (c < 64) out0[(size_t)t * H_SZ + nh * 64 + d] = o * 0.125f;  // q, pre-scaled
        else        out1[(size_t)t * H_SZ + nh * 64 + d] = o;          // k
      }
    }
  }
}

// ---------------- RMSNorm(o2 + residual) * w ----------------
__global__ void rmsnorm_add(const float* __restrict__ o2, const float* __restrict__ x,
                            const float* __restrict__ w, float* __restrict__ out) {
  int t = blockIdx.x;
  __shared__ float red[256];
  const float* a = o2 + (size_t)t * H_SZ;
  const float* b = x  + (size_t)t * H_SZ;
  float ss = 0.0f;
  for (int i = threadIdx.x; i < H_SZ; i += 256) { float v = a[i] + b[i]; ss += v * v; }
  red[threadIdx.x] = ss; __syncthreads();
  for (int s2 = 128; s2 > 0; s2 >>= 1) {
    if (threadIdx.x < s2) red[threadIdx.x] += red[threadIdx.x + s2];
    __syncthreads();
  }
  float r = rsqrtf(red[0] / (float)H_SZ + 1e-6f);
  for (int i = threadIdx.x; i < H_SZ; i += 256) {
    float v = a[i] + b[i];
    out[(size_t)t * H_SZ + i] = v * r * w[i];
  }
}

// ---------------- flash attention: 1 wave per (b, head, 16-query tile) ----------------
__global__ void __launch_bounds__(32)
flash_attn(const float* __restrict__ q, const float* __restrict__ k,
           const float* __restrict__ vT, float* __restrict__ ctx) {
  const int lane = threadIdx.x;
  const int qt = blockIdx.x, nh = blockIdx.y, b = blockIdx.z;
  const int l15 = lane & 15;
  const int half16 = (lane & 16) ? 16 : 0;
  const int base8 = (lane & 16) ? 8 : 0;

  // Q fragments (16 rows x 64 d), hi/lo, score scale folded into q already
  v16bf qhi[2], qlo[2];
  {
    const float* qp = q + (size_t)(b * S_SZ + qt * 16 + l15) * H_SZ + nh * 64;
    #pragma unroll
    for (int dt = 0; dt < 2; ++dt) {
      const float* ap = qp + dt * 32 + half16;
      #pragma unroll
      for (int i = 0; i < 8; ++i) {
        float f0 = ap[i], f1 = ap[8 + i];
        __bf16 h0 = (__bf16)f0, h1 = (__bf16)f1;
        qhi[dt][i] = h0; qhi[dt][8 + i] = h1;
        qlo[dt][i] = (__bf16)(f0 - (float)h0);
        qlo[dt][8 + i] = (__bf16)(f1 - (float)h1);
      }
    }
  }

  float mrun[8], lrun[8];
  v8f o[4];
  #pragma unroll
  for (int r = 0; r < 8; ++r) { mrun[r] = -3.0e30f; lrun[r] = 0.0f; }
  #pragma unroll
  for (int j = 0; j < 4; ++j) o[j] = (v8f)(0.0f);

  __shared__ __attribute__((aligned(32))) __bf16 Plds[512];  // P in A-frag layout

  const int lastk = qt * 16 + 15;
  const int nkb = (lastk >> 5) + 1;

  for (int kb = 0; kb < nkb; ++kb) {
    const int k0 = kb * 32;
    v8f s0 = (v8f)(0.0f), s1 = (v8f)(0.0f);

    // scores: A=q(16x64), B=k^T fragments (d-major rows, key columns)
    #pragma unroll
    for (int dt = 0; dt < 2; ++dt) {
      int doff = dt * 32 + half16;
      const float* kp0 = k + (size_t)(b * S_SZ + k0 + l15) * H_SZ + nh * 64 + doff;
      const float* kp1 = kp0 + (size_t)16 * H_SZ;
      v16bf k0hi, k0lo, k1hi, k1lo;
      #pragma unroll
      for (int e = 0; e < 16; ++e) {
        float f0 = kp0[e], f1 = kp1[e];
        __bf16 h0 = (__bf16)f0, h1 = (__bf16)f1;
        k0hi[e] = h0; k0lo[e] = (__bf16)(f0 - (float)h0);
        k1hi[e] = h1; k1lo[e] = (__bf16)(f1 - (float)h1);
      }
      s0 = WMMA_BF16(qhi[dt], k0hi, s0);
      s0 = WMMA_BF16(qhi[dt], k0lo, s0);
      s0 = WMMA_BF16(qlo[dt], k0hi, s0);
      s1 = WMMA_BF16(qhi[dt], k1hi, s1);
      s1 = WMMA_BF16(qhi[dt], k1lo, s1);
      s1 = WMMA_BF16(qlo[dt], k1hi, s1);
    }

    const bool needmask = (k0 + 31 > qt * 16);
    float fac[8];
    #pragma unroll
    for (int r = 0; r < 8; ++r) {
      if (needmask) {
        int qrow = qt * 16 + base8 + r;
        if (k0 + l15 > qrow)      s0[r] = -3.0e38f;
        if (k0 + 16 + l15 > qrow) s1[r] = -3.0e38f;
      }
      float mx = fmaxf(s0[r], s1[r]);
      #pragma unroll
      for (int msk = 8; msk >= 1; msk >>= 1) mx = fmaxf(mx, __shfl_xor(mx, msk, 32));
      float mn = fmaxf(mrun[r], mx);
      fac[r] = __expf(mrun[r] - mn);
      mrun[r] = mn;
      float p0 = __expf(s0[r] - mn);
      float p1 = __expf(s1[r] - mn);
      s0[r] = p0; s1[r] = p1;
      float rs = p0 + p1;
      #pragma unroll
      for (int msk = 8; msk >= 1; msk >>= 1) rs += __shfl_xor(rs, msk, 32);
      lrun[r] = lrun[r] * fac[r] + rs;
    }
    #pragma unroll
    for (int j = 0; j < 4; ++j)
      #pragma unroll
      for (int r = 0; r < 8; ++r) o[j][r] *= fac[r];

    // P (C-frag) -> LDS in A-frag layout: (M,kk) -> lane M+((kk&16)?16:0), e=kk&15
    #pragma unroll
    for (int r = 0; r < 8; ++r) {
      int M = base8 + r;
      int kk0 = l15;
      Plds[M * 16 + kk0] = (__bf16)s0[r];
      int kk1 = 16 + l15;
      Plds[(M + 16) * 16 + (kk1 & 15)] = (__bf16)s1[r];
    }
    __syncthreads();
    v16bf pa = *(const v16bf*)&Plds[lane * 16];

    // O += P(16x32) * V(32 keys x 64 d), V stored d-major so key runs are contiguous
    #pragma unroll
    for (int j = 0; j < 4; ++j) {
      int hd = j * 16 + l15;
      const float* vp = vT + ((size_t)(b * NH_SZ + nh) * HD_SZ + hd) * S_SZ + k0 + half16;
      v16bf vhi, vlo;
      #pragma unroll
      for (int e = 0; e < 16; ++e) {
        float f = vp[e];
        __bf16 h = (__bf16)f;
        vhi[e] = h; vlo[e] = (__bf16)(f - (float)h);
      }
      o[j] = WMMA_BF16(pa, vhi, o[j]);
      o[j] = WMMA_BF16(pa, vlo, o[j]);
    }
    __syncthreads();
  }

  #pragma unroll
  for (int j = 0; j < 4; ++j)
    #pragma unroll
    for (int r = 0; r < 8; ++r) {
      int qrow = qt * 16 + base8 + r;
      ctx[(size_t)(b * S_SZ + qrow) * H_SZ + nh * 64 + j * 16 + l15] = o[j][r] / lrun[r];
    }
}

// ---------------- host-side launcher ----------------
extern "C" void kernel_launch(void* const* d_in, const int* in_sizes, int n_in,
                              void* d_out, int out_size, void* d_ws, size_t ws_size,
                              hipStream_t stream) {
  (void)in_sizes; (void)n_in; (void)out_size; (void)ws_size;
  const float* x    = (const float*)d_in[0];
  const int*   pos  = (const int*)  d_in[1];
  const float* lf1c = (const float*)d_in[2];
  const float* lf2c = (const float*)d_in[3];
  const float* Wqk  = (const float*)d_in[4];
  const float* Wv   = (const float*)d_in[5];
  const float* Wo   = (const float*)d_in[6];
  const float* c1w  = (const float*)d_in[7];
  const float* c1b  = (const float*)d_in[8];
  const float* c2w  = (const float*)d_in[9];
  const float* c2b  = (const float*)d_in[10];
  const float* lnw  = (const float*)d_in[11];
  float* out = (float*)d_out;

  // ---- workspace arena (statically aliased; ~240 MiB peak) ----
  char* w = (char*)d_ws;
  float*  xcat  = (float*)(w + 0);           // 64 MiB  (T x 4096)
  float*  o1cat = (float*)(w + 0);           // alias: xcat dead after conv1 GEMM
  float*  o2    = (float*)(w + 33554432);    // 32 MiB
  float*  ctx   = (float*)(w + 0);           // alias: o1cat/o2 dead after rmsnorm
  float*  o1    = (float*)(w + 67108864);    // 16 MiB
  float*  lf    = (float*)(w + 83886080);    // 32 MiB
  float*  qrot  = (float*)(w + 117440512);   // 32 MiB
  float*  krot  = (float*)(w + 150994944);   // 32 MiB
  float*  vT    = (float*)(w + 184549376);   // 32 MiB
  __bf16* Whi   = (__bf16*)(w + 218103808);  // 16 MiB (reused across all 5 GEMMs)
  __bf16* Wlo   = (__bf16*)(w + 234881024);  // 16 MiB

  const dim3 blk32(32), blk128(128), blk256(256);
  const int MT = T_SZ / 16;  // 256 row tiles

  // --- G1: V = X * Wv^T  -> vT (d-major per head) ---
  pack_b<<<dim3((128 * 64 * 32 + 255) / 256), blk256, 0, stream>>>(
      Wv, 128, 64, H_SZ, 0, 0, Whi, Wlo);
  gemm_bf16x3<<<dim3(H_SZ / 256, MT), blk128, 0, stream>>>(
      x, H_SZ, H_SZ, Whi, Wlo, nullptr, 1, vT, 0, nullptr, nullptr);

  // --- conv1 as GEMM over [x[s-1]|x[s]] ---
  build_cat<<<dim3((unsigned)(((long)T_SZ * 2 * H_SZ + 255) / 256)), blk256, 0, stream>>>(
      x, lf1c, H_SZ, xcat);
  pack_b<<<dim3((64 * 128 * 32 + 255) / 256), blk256, 0, stream>>>(
      c1w, 64, 128, 2 * H_SZ, 1, H_SZ, Whi, Wlo);
  gemm_bf16x3<<<dim3(1024 / 256, MT), blk128, 0, stream>>>(
      xcat, 4096, 4096, Whi, Wlo, c1b, 0, o1, 1024, nullptr, nullptr);

  // --- conv2 as GEMM over [o1[s-1]|o1[s]] ---
  build_cat<<<dim3((unsigned)(((long)T_SZ * 2048 + 255) / 256)), blk256, 0, stream>>>(
      o1, lf2c, 1024, o1cat);
  pack_b<<<dim3((128 * 64 * 32 + 255) / 256), blk256, 0, stream>>>(
      c2w, 128, 64, 2048, 1, 1024, Whi, Wlo);
  gemm_bf16x3<<<dim3(H_SZ / 256, MT), blk128, 0, stream>>>(
      o1cat, 2048, 2048, Whi, Wlo, c2b, 0, o2, H_SZ, nullptr, nullptr);

  // --- lf = rmsnorm(o2 + x) * ln_w ---
  rmsnorm_add<<<dim3(T_SZ), blk256, 0, stream>>>(o2, x, lnw, lf);

  // --- G4: QK = lf * Wqk^T with fused YaRN RoPE (q scaled by 1/8) ---
  pack_b<<<dim3((256 * 64 * 32 + 255) / 256), blk256, 0, stream>>>(
      Wqk, 256, 64, H_SZ, 0, 0, Whi, Wlo);
  gemm_bf16x3<<<dim3(4096 / 256, MT), blk128, 0, stream>>>(
      lf, H_SZ, H_SZ, Whi, Wlo, nullptr, 2, qrot, 0, krot, pos);

  // --- causal flash attention ---
  flash_attn<<<dim3(S_SZ / 16, NH_SZ, B_SZ), blk32, 0, stream>>>(qrot, krot, vT, ctx);

  // --- G5: out = ctx * Wo^T ---
  pack_b<<<dim3((128 * 64 * 32 + 255) / 256), blk256, 0, stream>>>(
      Wo, 128, 64, H_SZ, 0, 0, Whi, Wlo);
  gemm_bf16x3<<<dim3(H_SZ / 256, MT), blk128, 0, stream>>>(
      ctx, H_SZ, H_SZ, Whi, Wlo, nullptr, 0, out, H_SZ, nullptr, nullptr);
}
